// FastASTEncoder_27410481283476
// MI455X (gfx1250) — matrix operand order repaired and checked by reading, not compiled, not run on hardware
//
#include <hip/hip_runtime.h>
#include <cstdint>
#include <cstddef>

// ---------------------------------------------------------------------------
// Problem constants (match reference)
// ---------------------------------------------------------------------------
#define NLAY 4
#define BB   4
#define LL   2048
#define DD   512
#define HH   8
#define PP   8
#define FF   2048
#define DKK  64
#define MM   (BB * LL)      /* 8192 rows of the token matrix */
#define HA_  4              /* ancestor heads */
#define SCALE_ 0.125f       /* 1/sqrt(64) */

typedef __attribute__((ext_vector_type(16))) _Float16 v16h;
typedef __attribute__((ext_vector_type(8)))  float    v8f;

// ---------------------------------------------------------------------------
// Helpers
// ---------------------------------------------------------------------------
__device__ __forceinline__ float gelu_tanh(float x) {
    // jax.nn.gelu default (approximate=True)
    float x3 = x * x * x;
    return 0.5f * x * (1.0f + tanhf(0.7978845608028654f * (x + 0.044715f * x3)));
}

union AFrag { v16h v; uint4 q[2]; };

// ---------------------------------------------------------------------------
// Weight convert + transpose:  Wt[n*K + k] = (f16) W[k*N + n]
// ---------------------------------------------------------------------------
__global__ void conv_transpose_f16(const float* __restrict__ W,
                                   _Float16* __restrict__ Wt,
                                   int K, int N) {
    int idx = blockIdx.x * blockDim.x + threadIdx.x;
    if (idx >= K * N) return;
    int n = idx / K;
    int k = idx - n * K;
    Wt[idx] = (_Float16)W[(size_t)k * N + n];
}

// ---------------------------------------------------------------------------
// LayerNorm (fp32 in) -> f16 out.  One 256-thread block per row, D = 512.
// ---------------------------------------------------------------------------
__global__ __launch_bounds__(256) void ln_to_f16(const float* __restrict__ x,
                                                 const float* __restrict__ g,
                                                 const float* __restrict__ b,
                                                 _Float16* __restrict__ y) {
    __shared__ float red[256];
    const int t = threadIdx.x;
    const size_t base = (size_t)blockIdx.x * DD;
    float v0 = x[base + t];
    float v1 = x[base + t + 256];
    red[t] = v0 + v1;
    __syncthreads();
    for (int s = 128; s > 0; s >>= 1) {
        if (t < s) red[t] += red[t + s];
        __syncthreads();
    }
    const float mu = red[0] * (1.0f / DD);
    __syncthreads();
    float d0 = v0 - mu, d1 = v1 - mu;
    red[t] = d0 * d0 + d1 * d1;
    __syncthreads();
    for (int s = 128; s > 0; s >>= 1) {
        if (t < s) red[t] += red[t + s];
        __syncthreads();
    }
    const float rs = rsqrtf(red[0] * (1.0f / DD) + 1e-5f);
    y[base + t]       = (_Float16)(d0 * rs * g[t]       + b[t]);
    y[base + t + 256] = (_Float16)(d1 * rs * g[t + 256] + b[t + 256]);
}

// Final LayerNorm, fp32 in-place on d_out
__global__ __launch_bounds__(256) void ln_final_f32(float* __restrict__ x,
                                                    const float* __restrict__ g,
                                                    const float* __restrict__ b) {
    __shared__ float red[256];
    const int t = threadIdx.x;
    const size_t base = (size_t)blockIdx.x * DD;
    float v0 = x[base + t];
    float v1 = x[base + t + 256];
    red[t] = v0 + v1;
    __syncthreads();
    for (int s = 128; s > 0; s >>= 1) {
        if (t < s) red[t] += red[t + s];
        __syncthreads();
    }
    const float mu = red[0] * (1.0f / DD);
    __syncthreads();
    float d0 = v0 - mu, d1 = v1 - mu;
    red[t] = d0 * d0 + d1 * d1;
    __syncthreads();
    for (int s = 128; s > 0; s >>= 1) {
        if (t < s) red[t] += red[t + s];
        __syncthreads();
    }
    const float rs = rsqrtf(red[0] * (1.0f / DD) + 1e-5f);
    x[base + t]       = d0 * rs * g[t]       + b[t];
    x[base + t + 256] = d1 * rs * g[t + 256] + b[t + 256];
}

// ---------------------------------------------------------------------------
// WMMA GEMM:  C[M,N] = A[M,K](f16, row-major) x Wt[N,K](f16, K-contiguous) + bias
// One wave per 64x64 output tile: 4x4 blocking of 16x16x32 WMMA tiles.
//   mode 0: outH = f16(acc + bias)              (QKV projection)
//   mode 1: outH = f16(gelu(acc + bias))        (FFN up-projection)
//   mode 2: outF += acc + bias                  (residual-accumulating proj)
//
// A-fragment layout (16-bit A 16x32): lane m = lane&15,
//   half-sel h = lane>>4; element e <-> K = h*16 + e (same order as B)
//   => two contiguous 16-byte loads per lane.
// B-fragment (32x16): lane n = lane&15, element e <-> K = h*16 + e
//   => with Wt stored [N,K], two contiguous 16-byte loads per lane.
// ---------------------------------------------------------------------------
__global__ __launch_bounds__(256) void gemm_wmma_f16(
    const _Float16* __restrict__ A, const _Float16* __restrict__ Wt,
    const float* __restrict__ bias, int M, int N, int K, int mode,
    _Float16* __restrict__ outH, float* __restrict__ outF) {

    const int lane = threadIdx.x & 31;
    const int wave = (blockIdx.x * blockDim.x + threadIdx.x) >> 5;
    const int nt = N >> 6;                  // 64-wide N tiles
    const int tM = (wave / nt) << 6;
    const int tN = (wave - (wave / nt) * nt) << 6;
    const int row  = lane & 15;
    const int hsel = lane >> 4;

    const _Float16* aP[4];
    const _Float16* bP[4];
#pragma unroll
    for (int i = 0; i < 4; ++i)
        aP[i] = A + (size_t)(tM + i * 16 + row) * K + hsel * 16;
#pragma unroll
    for (int j = 0; j < 4; ++j)
        bP[j] = Wt + (size_t)(tN + j * 16 + row) * K + hsel * 16;

    v8f acc[4][4] = {};

    for (int k0 = 0; k0 < K; k0 += 32) {
        AFrag a[4], b[4];
#pragma unroll
        for (int i = 0; i < 4; ++i) {
            a[i].q[0] = *(const uint4*)(aP[i] + k0);
            a[i].q[1] = *(const uint4*)(aP[i] + k0 + 8);
        }
#pragma unroll
        for (int j = 0; j < 4; ++j) {
            b[j].q[0] = *(const uint4*)(bP[j] + k0);
            b[j].q[1] = *(const uint4*)(bP[j] + k0 + 8);
        }
        if (k0 + 32 < K) {   // prefetch next k-step (global_prefetch_b8)
#pragma unroll
            for (int i = 0; i < 4; ++i) __builtin_prefetch(aP[i] + k0 + 32, 0, 3);
#pragma unroll
            for (int j = 0; j < 4; ++j) __builtin_prefetch(bP[j] + k0 + 32, 0, 3);
        }
#pragma unroll
        for (int i = 0; i < 4; ++i)
#pragma unroll
            for (int j = 0; j < 4; ++j)
                acc[i][j] = __builtin_amdgcn_wmma_f32_16x16x32_f16(
                    false, a[i].v, false, b[j].v, (short)0, acc[i][j],
                    false, false);
    }

    // Epilogue.  C/D layout: VGPR r -> M = r + (lane>>4)*8, N = lane&15.
#pragma unroll
    for (int j = 0; j < 4; ++j) {
        const int n = tN + j * 16 + row;
        const float bs = bias[n];
#pragma unroll
        for (int i = 0; i < 4; ++i) {
            const int mbase = tM + i * 16 + hsel * 8;
#pragma unroll
            for (int r = 0; r < 8; ++r) {
                float val = acc[i][j][r] + bs;
                size_t off = (size_t)(mbase + r) * N + n;
                if (mode == 0)      outH[off] = (_Float16)val;
                else if (mode == 1) outH[off] = (_Float16)gelu_tanh(val);
                else                outF[off] += val;
            }
        }
    }
}

// ---------------------------------------------------------------------------
// Tree-relative gather attention.  One wave32 per (b, h, l); P=8 edges,
// DK=64 -> 2 dims per lane; wave shuffle reductions for the dot products.
// qkv is f16 [M, 3D] (q @0, k @D, v @2D within a row).
// ---------------------------------------------------------------------------
__global__ __launch_bounds__(256) void tree_attn(
    const _Float16* __restrict__ qkv,
    const int* __restrict__ anc, const int* __restrict__ sib,
    const float* __restrict__ rel_q, const float* __restrict__ rel_k,
    const float* __restrict__ rel_v, _Float16* __restrict__ outH) {

    const int wave = (blockIdx.x * blockDim.x + threadIdx.x) >> 5;
    const int lane = threadIdx.x & 31;
    // wave = (b*H + h)*L + l
    const int l  = wave & (LL - 1);
    const int bh = wave >> 11;       // / L
    const int h  = bh & (HH - 1);
    const int b  = bh >> 3;

    const int d0 = lane * 2;
    const size_t qoff = ((size_t)(b * LL + l)) * (3 * DD) + h * DKK + d0;
    const float q0 = (float)qkv[qoff];
    const float q1 = (float)qkv[qoff + 1];

    const int* edges = (h < HA_) ? anc : sib;
    const int ebase = b * (PP * LL) + l;

    float s[PP], vg0[PP], vg1[PP];
#pragma unroll
    for (int p = 0; p < PP; ++p) {
        const int e = edges[ebase + p * LL];
        const bool valid = (e >= 0);
        const int ei = valid ? e : 0;
        const size_t koff =
            ((size_t)(b * LL + ei)) * (3 * DD) + DD + h * DKK + d0;
        const float k0 = (float)qkv[koff];
        const float k1 = (float)qkv[koff + 1];
        vg0[p] = (float)qkv[koff + DD];          // v is D further along the row
        vg1[p] = (float)qkv[koff + DD + 1];
        const int rb = (h * PP + p) * DKK + d0;
        const float rq0 = rel_q[rb], rq1 = rel_q[rb + 1];
        const float rk0 = rel_k[rb], rk1 = rel_k[rb + 1];
        // q.kg + rel_q.kg + q.rel_k
        float part = (q0 + rq0) * k0 + (q1 + rq1) * k1 + q0 * rk0 + q1 * rk1;
#pragma unroll
        for (int off = 16; off > 0; off >>= 1)
            part += __shfl_xor(part, off, 32);
        s[p] = valid ? part * SCALE_ : -1e9f;
    }

    // softmax over P (replicated across lanes)
    float mx = s[0];
#pragma unroll
    for (int p = 1; p < PP; ++p) mx = fmaxf(mx, s[p]);
    float w[PP], sw = 0.0f;
#pragma unroll
    for (int p = 0; p < PP; ++p) { w[p] = __expf(s[p] - mx); sw += w[p]; }
    const float inv = 1.0f / sw;

    float a0 = 0.0f, a1 = 0.0f;
#pragma unroll
    for (int p = 0; p < PP; ++p) {
        const int rb = (h * PP + p) * DKK + d0;
        a0 += w[p] * (vg0[p] + rel_v[rb]);
        a1 += w[p] * (vg1[p] + rel_v[rb + 1]);
    }
    const size_t ooff = ((size_t)(b * LL + l)) * DD + h * DKK + d0;
    outH[ooff]     = (_Float16)(a0 * inv);
    outH[ooff + 1] = (_Float16)(a1 * inv);
}

// ---------------------------------------------------------------------------
// Launcher
// ---------------------------------------------------------------------------
extern "C" void kernel_launch(void* const* d_in, const int* in_sizes, int n_in,
                              void* d_out, int out_size, void* d_ws,
                              size_t ws_size, hipStream_t stream) {
    (void)in_sizes; (void)n_in; (void)out_size; (void)ws_size;

    const float* src_emb = (const float*)d_in[0];
    const int*   anc     = (const int*)d_in[1];
    const int*   sib     = (const int*)d_in[2];
    const float* Wqkv    = (const float*)d_in[3];
    const float* bqkv    = (const float*)d_in[4];
    const float* Wo      = (const float*)d_in[5];
    const float* bo      = (const float*)d_in[6];
    const float* ln1_g   = (const float*)d_in[7];
    const float* ln1_b   = (const float*)d_in[8];
    const float* ln2_g   = (const float*)d_in[9];
    const float* ln2_b   = (const float*)d_in[10];
    const float* W1      = (const float*)d_in[11];
    const float* b1      = (const float*)d_in[12];
    const float* W2      = (const float*)d_in[13];
    const float* b2      = (const float*)d_in[14];
    const float* lnf_g   = (const float*)d_in[15];
    const float* lnf_b   = (const float*)d_in[16];
    const float* rel_q   = (const float*)d_in[17];
    const float* rel_k   = (const float*)d_in[18];
    const float* rel_v   = (const float*)d_in[19];

    float* out = (float*)d_out;   // residual stream lives in d_out (fp32)

    // ---- workspace layout (f16 halves) ----
    char* ws = (char*)d_ws;
    size_t off = 0;
    auto carve = [&](size_t halves) {
        _Float16* p = (_Float16*)(ws + off);
        off += ((halves * sizeof(_Float16)) + 255) & ~(size_t)255;
        return p;
    };
    _Float16* Wt_qkv = carve((size_t)NLAY * 3 * DD * DD);   // [NL][1536][512]
    _Float16* Wt_o   = carve((size_t)NLAY * DD * DD);       // [NL][512][512]
    _Float16* Wt_1   = carve((size_t)NLAY * FF * DD);       // [NL][2048][512]
    _Float16* Wt_2   = carve((size_t)NLAY * DD * FF);       // [NL][512][2048]
    _Float16* xln    = carve((size_t)MM * DD);              // LN output / FFN in
    _Float16* attn   = carve((size_t)MM * DD);              // attention output
    _Float16* big    = carve((size_t)MM * FF);              // qkv (M x 1536) then g (M x 2048)
    _Float16* qkv16  = big;
    _Float16* gbuf   = big;

    // ---- convert + transpose weights to f16 [N,K] ----
    auto conv = [&](const float* W, _Float16* Wt, int K, int N) {
        int elems = K * N;
        conv_transpose_f16<<<(elems + 255) / 256, 256, 0, stream>>>(W, Wt, K, N);
    };
    for (int i = 0; i < NLAY; ++i) {
        conv(Wqkv + (size_t)i * DD * 3 * DD, Wt_qkv + (size_t)i * 3 * DD * DD, DD, 3 * DD);
        conv(Wo   + (size_t)i * DD * DD,     Wt_o   + (size_t)i * DD * DD,     DD, DD);
        conv(W1   + (size_t)i * DD * FF,     Wt_1   + (size_t)i * FF * DD,     DD, FF);
        conv(W2   + (size_t)i * FF * DD,     Wt_2   + (size_t)i * DD * FF,     FF, DD);
    }

    // ---- residual stream init ----
    hipMemcpyAsync(out, src_emb, (size_t)MM * DD * sizeof(float),
                   hipMemcpyDeviceToDevice, stream);

    auto gemm = [&](const _Float16* A, const _Float16* Wt, const float* bias,
                    int N, int K, int mode, _Float16* oh, float* of) {
        int waves = (MM / 64) * (N / 64);        // always a multiple of 8
        gemm_wmma_f16<<<waves / 8, 256, 0, stream>>>(A, Wt, bias, MM, N, K,
                                                     mode, oh, of);
    };

    for (int i = 0; i < NLAY; ++i) {
        // sublayer 0: pre-LN tree-relative attention
        ln_to_f16<<<MM, 256, 0, stream>>>(out, ln1_g + i * DD, ln1_b + i * DD, xln);
        gemm(xln, Wt_qkv + (size_t)i * 3 * DD * DD, bqkv + (size_t)i * 3 * DD,
             3 * DD, DD, /*mode=*/0, qkv16, nullptr);
        {
            int waves = BB * HH * LL;            // 65536 waves, 8 per block
            tree_attn<<<waves / 8, 256, 0, stream>>>(qkv16, anc, sib,
                                                     rel_q, rel_k, rel_v, attn);
        }
        gemm(attn, Wt_o + (size_t)i * DD * DD, bo + (size_t)i * DD,
             DD, DD, /*mode=*/2, nullptr, out);

        // sublayer 1: pre-LN feed-forward with GELU
        ln_to_f16<<<MM, 256, 0, stream>>>(out, ln2_g + i * DD, ln2_b + i * DD, xln);
        gemm(xln, Wt_1 + (size_t)i * FF * DD, b1 + (size_t)i * FF,
             FF, DD, /*mode=*/1, gbuf, nullptr);
        gemm(gbuf, Wt_2 + (size_t)i * DD * FF, b2 + (size_t)i * DD,
             DD, FF, /*mode=*/2, nullptr, out);
    }

    ln_final_f32<<<MM, 256, 0, stream>>>(out, lnf_g, lnf_b);
}
